// LinDBLayer_55585466745382
// MI455X (gfx1250) — matrix operand order repaired and checked by reading, not compiled, not run on hardware
//
#include <hip/hip_runtime.h>

// ---- types ----------------------------------------------------------------
typedef __bf16 v16bf __attribute__((ext_vector_type(16)));
typedef __bf16 v8bf  __attribute__((ext_vector_type(8)));
typedef float  v8f   __attribute__((ext_vector_type(8)));

#define CCH  128   // channels
#define KTOT 256   // fused K = [dv | e] or [agg | x]
#define APAD (KTOT + 8)

// ---------------------------------------------------------------------------
// Prep: build transposed fused bf16 weights in workspace.
//   WtE[n][k] : k<128 -> W_en[k][n] ; k>=128 -> beta_e[n][k-128] + I
//   WtN[n][k] : k<128 -> W_ne[k][n] ; k>=128 -> beta_n[n][k-128] + I
// ---------------------------------------------------------------------------
__global__ void prep_weights(const float* __restrict__ W_ne,
                             const float* __restrict__ W_en,
                             const float* __restrict__ beta_e,
                             const float* __restrict__ beta_n,
                             __bf16* __restrict__ WtE,
                             __bf16* __restrict__ WtN) {
  int idx = blockIdx.x * blockDim.x + threadIdx.x;
  if (idx >= CCH * KTOT) return;
  int n = idx >> 8;          // column of the GEMM (0..127)
  int k = idx & (KTOT - 1);  // fused K index (0..255)
  float ve, vn;
  if (k < CCH) {
    ve = W_en[k * CCH + n];
    vn = W_ne[k * CCH + n];
  } else {
    int r = k - CCH;
    float id = (r == n) ? 1.0f : 0.0f;
    ve = beta_e[n * CCH + r] + id;
    vn = beta_n[n * CCH + r] + id;
  }
  WtE[idx] = (__bf16)ve;
  WtN[idx] = (__bf16)vn;
}

__global__ void zero_f4(float4* __restrict__ p, int n4) {
  int i = blockIdx.x * blockDim.x + threadIdx.x;
  if (i < n4) p[i] = make_float4(0.f, 0.f, 0.f, 0.f);
}

// ---------------------------------------------------------------------------
// Shared compute phase: 16xKTOT bf16 tile (LDS) x WtG (global, transposed)
// -> 16x16 f32 per wave via v_wmma_f32_16x16x32_bf16, K accumulated = 256.
// ---------------------------------------------------------------------------
__device__ __forceinline__ v8f wmma_tile_k256(const __bf16 (*Ash)[APAD],
                                              const __bf16* __restrict__ Wt,
                                              int lane, int n0) {
  const int  col = lane & 15;
  const bool hi  = (lane & 16) != 0;
  v8f acc = {};
#pragma unroll
  for (int k0 = 0; k0 < KTOT; k0 += 32) {
    // A fragment (contiguous K, matching B): lanes 0-15: K k0..k0+15
    //                                         lanes16-31: K k0+16..k0+31
    const int kl = k0 + (hi ? 16 : 0);
    v8bf alo = *(const v8bf*)&Ash[col][kl];
    v8bf ahi = *(const v8bf*)&Ash[col][kl + 8];
    v16bf a = __builtin_shufflevector(alo, ahi, 0, 1, 2, 3, 4, 5, 6, 7,
                                      8, 9, 10, 11, 12, 13, 14, 15);
    // B fragment: lane = column n0+col; lanes 0-15 hold K k0..k0+15,
    // lanes 16-31 hold K k0+16..k0+31 -> contiguous in transposed weights.
    const int kb = k0 + (hi ? 16 : 0);
    v16bf b = *(const v16bf*)&Wt[(n0 + col) * KTOT + kb];
    acc = __builtin_amdgcn_wmma_f32_16x16x32_bf16(
        false, a, false, b, (short)0, acc, false, false);
  }
  return acc;
}

// ---------------------------------------------------------------------------
// Edge kernel: tile of 16 edges per block (256 thr = 8 waves).
//   e_new = relu([v_i - v_j | e] @ WtE)   (written to e-section of d_out)
//   agg  += scatter_add(e by target)      (f32 atomics into x-section)
// ---------------------------------------------------------------------------
__global__ __launch_bounds__(256) void edge_kernel(
    const float* __restrict__ x, const int* __restrict__ eidx,
    const float* __restrict__ e, const __bf16* __restrict__ WtE,
    float* __restrict__ e_out, float* __restrict__ agg, int E) {
  __shared__ __bf16 Ash[16][APAD];
  const int t0  = blockIdx.x * 16;
  const int tid = threadIdx.x;

  // ---- staging: gather, diff, bf16-pack, scatter-add -----------------------
  {
    const int m  = tid >> 4;
    const int c0 = (tid & 15) * 8;
    const int eg = t0 + m;
    const int src = eidx[eg];        // edge_index[0][eg]
    const int dst = eidx[E + eg];    // edge_index[1][eg]
    const float4 xs0 = *(const float4*)&x[(size_t)src * CCH + c0];
    const float4 xs1 = *(const float4*)&x[(size_t)src * CCH + c0 + 4];
    const float4 xd0 = *(const float4*)&x[(size_t)dst * CCH + c0];
    const float4 xd1 = *(const float4*)&x[(size_t)dst * CCH + c0 + 4];
    const float4 e0  = *(const float4*)&e[(size_t)eg * CCH + c0];
    const float4 e1  = *(const float4*)&e[(size_t)eg * CCH + c0 + 4];
    float dv[8] = {xd0.x - xs0.x, xd0.y - xs0.y, xd0.z - xs0.z, xd0.w - xs0.w,
                   xd1.x - xs1.x, xd1.y - xs1.y, xd1.z - xs1.z, xd1.w - xs1.w};
    float ev[8] = {e0.x, e0.y, e0.z, e0.w, e1.x, e1.y, e1.z, e1.w};
    v8bf dvb, evb;
#pragma unroll
    for (int i = 0; i < 8; ++i) {
      dvb[i] = (__bf16)dv[i];
      evb[i] = (__bf16)ev[i];
    }
    *(v8bf*)&Ash[m][c0]       = dvb;   // K 0..127  = v_i - v_j
    *(v8bf*)&Ash[m][CCH + c0] = evb;   // K 128..255 = e
    float* aggRow = &agg[(size_t)dst * CCH + c0];
#pragma unroll
    for (int i = 0; i < 8; ++i) atomicAdd(&aggRow[i], ev[i]);
  }
  __syncthreads();

  // ---- compute + relu + store ---------------------------------------------
  const int lane = tid & 31;
  const int n0   = (tid >> 5) * 16;
  v8f acc = wmma_tile_k256(Ash, WtE, lane, n0);
  const int  col = lane & 15;
  const int  mb  = (lane & 16) ? 8 : 0;  // C/D: v0-7 -> rows 0-7 / 8-15
#pragma unroll
  for (int r = 0; r < 8; ++r)
    e_out[(size_t)(t0 + mb + r) * CCH + n0 + col] = fmaxf(acc[r], 0.0f);
}

// ---------------------------------------------------------------------------
// Node kernel: tile of 16 nodes per block.
//   x_new = relu([agg | x] @ WtN), in-place over the agg buffer (x-section).
// ---------------------------------------------------------------------------
__global__ __launch_bounds__(256) void node_kernel(
    const float* __restrict__ x, const __bf16* __restrict__ WtN,
    float* __restrict__ xio /* agg in, x_new out */) {
  __shared__ __bf16 Ash[16][APAD];
  const int t0  = blockIdx.x * 16;
  const int tid = threadIdx.x;
  {
    const int m  = tid >> 4;
    const int c0 = (tid & 15) * 8;
    const int n  = t0 + m;
    const float4 a0 = *(const float4*)&xio[(size_t)n * CCH + c0];
    const float4 a1 = *(const float4*)&xio[(size_t)n * CCH + c0 + 4];
    const float4 x0 = *(const float4*)&x[(size_t)n * CCH + c0];
    const float4 x1 = *(const float4*)&x[(size_t)n * CCH + c0 + 4];
    float av[8] = {a0.x, a0.y, a0.z, a0.w, a1.x, a1.y, a1.z, a1.w};
    float xv[8] = {x0.x, x0.y, x0.z, x0.w, x1.x, x1.y, x1.z, x1.w};
    v8bf avb, xvb;
#pragma unroll
    for (int i = 0; i < 8; ++i) {
      avb[i] = (__bf16)av[i];
      xvb[i] = (__bf16)xv[i];
    }
    *(v8bf*)&Ash[m][c0]       = avb;   // K 0..127  = agg
    *(v8bf*)&Ash[m][CCH + c0] = xvb;   // K 128..255 = x
  }
  __syncthreads();

  const int lane = tid & 31;
  const int n0   = (tid >> 5) * 16;
  v8f acc = wmma_tile_k256(Ash, WtN, lane, n0);
  const int col = lane & 15;
  const int mb  = (lane & 16) ? 8 : 0;
#pragma unroll
  for (int r = 0; r < 8; ++r)
    xio[(size_t)(t0 + mb + r) * CCH + n0 + col] = fmaxf(acc[r], 0.0f);
}

// ---------------------------------------------------------------------------
extern "C" void kernel_launch(void* const* d_in, const int* in_sizes, int n_in,
                              void* d_out, int out_size, void* d_ws,
                              size_t ws_size, hipStream_t stream) {
  const float* x      = (const float*)d_in[0];
  const int*   eidx   = (const int*)d_in[1];
  const float* e      = (const float*)d_in[2];
  const float* W_ne   = (const float*)d_in[3];
  const float* W_en   = (const float*)d_in[4];
  const float* beta_e = (const float*)d_in[5];
  const float* beta_n = (const float*)d_in[6];

  const int N = in_sizes[0] / CCH;  // 50000
  const int E = in_sizes[2] / CCH;  // 800000

  float* x_out = (float*)d_out;                 // [N, C] (also agg scratch)
  float* e_out = x_out + (size_t)N * CCH;       // [E, C]

  __bf16* WtE = (__bf16*)d_ws;                  // [128][256] bf16 = 64 KB
  __bf16* WtN = WtE + CCH * KTOT;               // [128][256] bf16 = 64 KB

  // 1) fused/transposed bf16 weights
  prep_weights<<<(CCH * KTOT + 255) / 256, 256, 0, stream>>>(
      W_ne, W_en, beta_e, beta_n, WtE, WtN);
  // 2) zero agg accumulator (x-section of d_out)
  const int n4 = (N * CCH) / 4;
  zero_f4<<<(n4 + 255) / 256, 256, 0, stream>>>((float4*)x_out, n4);
  // 3) edge GEMM + scatter-add (E multiple of 16)
  edge_kernel<<<E / 16, 256, 0, stream>>>(x, eidx, e, WtE, e_out, x_out, E);
  // 4) node GEMM, in place (N multiple of 16)
  node_kernel<<<N / 16, 256, 0, stream>>>(x, WtN, x_out);
}
